// FlatNCE_32847909880225
// MI455X (gfx1250) — matrix operand order repaired and checked by reading, not compiled, not run on hardware
//
#include <hip/hip_runtime.h>

typedef __attribute__((ext_vector_type(16))) _Float16 v16h;
typedef __attribute__((ext_vector_type(8)))  _Float16 v8h;
typedef __attribute__((ext_vector_type(8)))  float    v8f;
typedef __attribute__((ext_vector_type(4)))  unsigned int v4u;
typedef __attribute__((ext_vector_type(8)))  int      v8i;
typedef __attribute__((ext_vector_type(4)))  int      v4i;

#define BATCH   4096
#define NROWS   8192          // 2*BATCH
#define DIM     256
#define INV_T   10.0f         // 1/0.1
#define KSTEPS  8             // 256 / 32
#define NCOLT   512           // 8192 / 16
#define WAVES   8
#define MROWS   32            // rows per wave (2 x 16-row A tiles)

#if defined(__has_builtin)
#if __has_builtin(__builtin_amdgcn_tensor_load_to_lds) && \
    __has_builtin(__builtin_amdgcn_s_wait_tensorcnt)
#define USE_TDM 1
#endif
#endif
#ifndef USE_TDM
#define USE_TDM 0
#endif

// ---------------------------------------------------------------------------
// Kernel 1: row L2-normalize (F.normalize semantics, eps clamp) -> fp16 f
// ---------------------------------------------------------------------------
__global__ __launch_bounds__(256) void
flatnce_normalize_kernel(const float* __restrict__ zi,
                         const float* __restrict__ zj,
                         _Float16* __restrict__ fH) {
  const int tid  = threadIdx.x;
  const int lane = tid & 31;
  const int wave = tid >> 5;
  const int row  = blockIdx.x * WAVES + wave;          // 1024 blocks * 8

  const float* src = (row < BATCH) ? (zi + (size_t)row * DIM)
                                   : (zj + (size_t)(row - BATCH) * DIM);
  const float4 a = *(const float4*)(src + lane * 8);
  const float4 b = *(const float4*)(src + lane * 8 + 4);

  float ss = a.x * a.x + a.y * a.y + a.z * a.z + a.w * a.w
           + b.x * b.x + b.y * b.y + b.z * b.z + b.w * b.w;
  #pragma unroll
  for (int m = 1; m <= 16; m <<= 1) ss += __shfl_xor(ss, m, 32);

  const float scale = 1.0f / fmaxf(sqrtf(ss), 1e-12f);

  v8h h;
  h[0] = (_Float16)(a.x * scale); h[1] = (_Float16)(a.y * scale);
  h[2] = (_Float16)(a.z * scale); h[3] = (_Float16)(a.w * scale);
  h[4] = (_Float16)(b.x * scale); h[5] = (_Float16)(b.y * scale);
  h[6] = (_Float16)(b.z * scale); h[7] = (_Float16)(b.w * scale);
  *(v8h*)(fH + (size_t)row * DIM + lane * 8) = h;
}

// ---------------------------------------------------------------------------
// Kernel 2: fused sim = f f^T (WMMA f16 -> f32 acc) + masked logsumexp.
// Each wave owns 32 rows (two A tiles resident in VGPRs); block double-buffers
// 16-column B tiles (8 KB) through LDS — staged by the Tensor Data Mover when
// the builtin is available, else by cooperative vector copies.
// ---------------------------------------------------------------------------
__global__ __launch_bounds__(256) void
flatnce_lse_kernel(const _Float16* __restrict__ fH, float* __restrict__ out) {
  __shared__ __align__(16) _Float16 tile[2][16 * DIM];   // 2 x 8 KB

  const int tid   = threadIdx.x;
  const int lane  = tid & 31;
  const int wave  = tid >> 5;
  const int half  = lane >> 4;          // 0: lanes 0-15, 1: lanes 16-31
  const int l15   = lane & 15;
  const int mBase = (blockIdx.x * WAVES + wave) * MROWS;   // 32 blocks

  // --- A operands: ISA 16-bit A 16x32 layout; rows mBase..mBase+31 resident.
  const int off = half ? 8 : 0;
  v16h aOp[2][KSTEPS];
  #pragma unroll
  for (int t = 0; t < 2; ++t) {
    const int aRow = mBase + t * 16 + l15;
    #pragma unroll
    for (int ks = 0; ks < KSTEPS; ++ks) {
      const _Float16* p = fH + (size_t)aRow * DIM + ks * 32 + off;
      const v8h lo = *(const v8h*)p;
      const v8h hi = *(const v8h*)(p + 16);
      #pragma unroll
      for (int i = 0; i < 8; ++i) { aOp[t][ks][i] = lo[i]; aOp[t][ks][i + 8] = hi[i]; }
    }
  }

#if USE_TDM
  // One wave per block drives the TDM: 2D D# for a 16-col x 512 B tile.
  auto tdm_load = [&](int ct, int buf) {
    // Generic LDS pointer low 32 bits == LDS byte offset (flat aperture rule).
    const unsigned lds = (unsigned)(unsigned long long)(void*)&tile[buf][0];
    const unsigned long long ga =
        (unsigned long long)(fH + (size_t)ct * 16 * DIM);
    v4u g0;
    g0[0] = 1u;                                   // count=1, user descriptor
    g0[1] = lds;                                  // lds_addr
    g0[2] = (unsigned)ga;                         // global_addr[31:0]
    g0[3] = (unsigned)((ga >> 32) & 0x01FFFFFFu)  // global_addr[56:32]
          | (2u << 30);                           // type=2 ("image")
    v8i g1;
    g1[0] = (int)(2u << 16);      // data_size=4B; mask=0; no pad/iterate
    g1[1] = (int)(128u << 16);    // tensor_dim0[15:0]=128 (4B units per col)
    g1[2] = (int)(8192u << 16);   // tensor_dim0[31:16]=0 | tensor_dim1[15:0]=8192
    g1[3] = (int)(128u << 16);    // tensor_dim1[31:16]=0 | tile_dim0=128
    g1[4] = 16;                   // tile_dim1=16 | tile_dim2=0
    g1[5] = 128;                  // tensor_dim0_stride[31:0]=128
    g1[6] = (int)(128u << 16);    // stride0[47:32]=0 | stride1[15:0]=128
    g1[7] = 0;                    // stride1[47:16]=0
    const v4i z4 = {};
#if __clang_major__ >= 23
    const v8i z8 = {};
    __builtin_amdgcn_tensor_load_to_lds(g0, g1, z4, z4, z8, 0);
#else
    __builtin_amdgcn_tensor_load_to_lds(g0, g1, z4, z4, 0);
#endif
  };
#else
  // Fallback: cooperative stage, 256 threads x 32 B.
  auto stage = [&](int ct, int buf) {
    const int c = tid >> 4;            // local column 0..15
    const int q = (tid & 15) * 2;      // int4 index within column (32 per col)
    const int4* g = (const int4*)fH + ((size_t)(ct * 16 + c) * (DIM / 8) + q);
    int4* s = (int4*)&tile[buf][0] + (c * (DIM / 8) + q);
    s[0] = g[0];
    s[1] = g[1];
  };
#endif

  float runs[2][8];   // per-lane partial sums of exp(sim/T)
  float posA[2][8];   // per-lane capture of sim[i, pos(i)]/T
  #pragma unroll
  for (int t = 0; t < 2; ++t)
    #pragma unroll
    for (int r = 0; r < 8; ++r) { runs[t][r] = 0.0f; posA[t][r] = 0.0f; }

  // Column tiles needing diag/positive masking for tile0 (tile1 uses +1).
  const int mtD = mBase >> 4;
  const int mtP = (mBase ^ BATCH) >> 4;

  auto acc_plain = [&](const v8f& c, float (&rs)[8]) {
    #pragma unroll
    for (int r = 0; r < 8; ++r) rs[r] += __expf(c[r] * INV_T);
  };
  auto acc_masked = [&](const v8f& c, float (&rs)[8], float (&ps)[8],
                        int rowBase, int gCol) {
    #pragma unroll
    for (int r = 0; r < 8; ++r) {
      const int gRow   = rowBase + half * 8 + r;
      const int posCol = gRow ^ BATCH;           // (i + B) % 2B
      const float x    = c[r] * INV_T;
      const bool  incl = (gCol != gRow) && (gCol != posCol);
      rs[r] += incl ? __expf(x) : 0.0f;
      ps[r] += (gCol == posCol) ? x : 0.0f;
    }
  };

#if USE_TDM
  if (wave == 0) tdm_load(0, 0);
#else
  stage(0, 0);
#endif

  for (int ct = 0; ct < NCOLT; ++ct) {
#if USE_TDM
    if (wave == 0) __builtin_amdgcn_s_wait_tensorcnt(0);
    __syncthreads();
    if (wave == 0 && ct + 1 < NCOLT) tdm_load(ct + 1, (ct + 1) & 1);
#else
    __syncthreads();
    if (ct + 1 < NCOLT) stage(ct + 1, (ct + 1) & 1);
    if (ct + 2 < NCOLT)
      __builtin_prefetch((const void*)((const int4*)fH +
                          (size_t)(ct + 2) * 16 * (DIM / 8) + (tid & 255)), 0, 1);
#endif

    const _Float16* tb = &tile[ct & 1][0];

    // Two 16x16 sim tiles; every B operand read feeds two WMMAs.
    v8f c0 = {}, c1 = {};
    #pragma unroll
    for (int ks = 0; ks < KSTEPS; ++ks) {
      // ISA 16-bit B 32x16 layout: lane l (col l%16) holds K 8h..8h+7 and 16+8h..16+8h+7.
      const _Float16* bp = tb + l15 * DIM + ks * 32 + (half ? 8 : 0);
      const v8h lo = *(const v8h*)bp;
      const v8h hi = *(const v8h*)(bp + 16);
      v16h bOp;
      #pragma unroll
      for (int i = 0; i < 8; ++i) { bOp[i] = lo[i]; bOp[i + 8] = hi[i]; }
      c0 = __builtin_amdgcn_wmma_f32_16x16x32_f16(
               false, aOp[0][ks], false, bOp, (short)0, c0, false, false);
      c1 = __builtin_amdgcn_wmma_f32_16x16x32_f16(
               false, aOp[1][ks], false, bOp, (short)0, c1, false, false);
    }

    const int gCol = ct * 16 + l15;
    if (ct == mtD || ct == mtP) acc_masked(c0, runs[0], posA[0], mBase, gCol);
    else                        acc_plain(c0, runs[0]);
    if (ct == mtD + 1 || ct == mtP + 1)
                                acc_masked(c1, runs[1], posA[1], mBase + 16, gCol);
    else                        acc_plain(c1, runs[1]);
  }

  // Final width-16 cross-lane reductions + output (C/D: VGPR r <-> row).
  #pragma unroll
  for (int t = 0; t < 2; ++t) {
    #pragma unroll
    for (int r = 0; r < 8; ++r) {
      float s = runs[t][r];
      float p = posA[t][r];
      #pragma unroll
      for (int m = 1; m <= 8; m <<= 1) {
        s += __shfl_xor(s, m, 16);
        p += __shfl_xor(p, m, 16);
      }
      const float clog = __logf(s) - p;            // clogits
      const float one  = __expf(clog - clog);      // reference forward value
      if (l15 == 0) out[mBase + t * 16 + half * 8 + r] = one;
    }
  }
}

extern "C" void kernel_launch(void* const* d_in, const int* in_sizes, int n_in,
                              void* d_out, int out_size, void* d_ws, size_t ws_size,
                              hipStream_t stream) {
  const float* zi = (const float*)d_in[0];   // [4096, 256] fp32
  const float* zj = (const float*)d_in[1];   // [4096, 256] fp32
  float* out = (float*)d_out;                // [8192] fp32
  _Float16* fH = (_Float16*)d_ws;            // 8192*256 fp16 = 4 MB scratch

  flatnce_normalize_kernel<<<NROWS / WAVES, 256, 0, stream>>>(zi, zj, fH);
  flatnce_lse_kernel<<<NROWS / (MROWS * WAVES), 256, 0, stream>>>(fH, out);
}
